// FormulaNet_77403900609202
// MI455X (gfx1250) — matrix operand order repaired and checked by reading, not compiled, not run on hardware
//
#include <hip/hip_runtime.h>
#include <hip/hip_bf16.h>

// Problem constants (from reference)
#define Dd   64
#define Nn   32000L
#define Ee   512000L
#define Gg   32
#define ITERS 2
#define EPSc 1e-5f
#define T1   4   // tiles (of 16 rows) per wave in the tiled GEMM kernels

typedef __attribute__((ext_vector_type(16))) _Float16 v16h;
typedef __attribute__((ext_vector_type(8)))  _Float16 v8h;
typedef __attribute__((ext_vector_type(8)))  float    v8f;
typedef __attribute__((ext_vector_type(4)))  float    v4f;

// ---------------------------------------------------------------------------
// Utility kernels
// ---------------------------------------------------------------------------
__global__ void zero_f32_kernel(float* __restrict__ p, long n) {
  long t = (long)blockIdx.x * blockDim.x + threadIdx.x;
  if (t < n) p[t] = 0.0f;
}

__global__ void init_out_kernel(float* __restrict__ p, long n) {
  long t = (long)blockIdx.x * blockDim.x + threadIdx.x;
  if (t < n) p[t] = -3.402823466e38f;
}

__global__ void nodes_init_kernel(const float* __restrict__ x,
                                  float* __restrict__ nodes,
                                  _Float16* __restrict__ nodes16, long n) {
  long t = (long)blockIdx.x * blockDim.x + threadIdx.x;
  if (t < n) { float v = x[t]; nodes[t] = v; nodes16[t] = (_Float16)v; }
}

// Pack a row-major [K,64] fp32 weight matrix into the WMMA B-operand lane
// layout (16-bit B 32x16 per tile): dst[((nb*KC + kc)*32 + lane)*16 + p]
// with lane = khi*16 + (n&15), K_in_chunk = khi*8 + (p&7) + (p>>3)*16.
__global__ void pack_w_kernel(const float* __restrict__ W,
                              _Float16* __restrict__ dst, int K) {
  int t = blockIdx.x * blockDim.x + threadIdx.x;
  if (t >= K * 64) return;
  int k = t / 64, n = t % 64;
  int kc = k >> 5, kl = k & 31, khi = (kl >> 3) & 1, p = (kl & 7) | ((kl >> 4) << 3);
  int nb = n >> 4, nl = n & 15;
  int lane = (khi << 4) | nl;
  int kChunks = K >> 5;
  dst[(((nb * kChunks + kc) * 32 + lane) << 4) + p] = (_Float16)W[k * 64 + n];
}

__global__ void deg_count_kernel(const int* __restrict__ idx,
                                 float* __restrict__ cnt, long E) {
  long t = (long)blockIdx.x * blockDim.x + threadIdx.x;
  if (t < E) atomicAdd(&cnt[idx[t]], 1.0f);
}

__global__ void deg_inv_kernel(float* __restrict__ d, long n) {
  long t = (long)blockIdx.x * blockDim.x + threadIdx.x;
  if (t < n) { float v = d[t]; d[t] = (v > 0.0f) ? 1.0f / v : 0.0f; }
}

// scale = g * rsqrt(var+eps), shift = be - mu*scale
__global__ void bn_finalize_kernel(const float* __restrict__ stats,
                                   const float* __restrict__ g,
                                   const float* __restrict__ be,
                                   float invCount, float* __restrict__ ss) {
  int c = threadIdx.x;  // blockDim = 64
  float mu  = stats[c] * invCount;
  float var = stats[64 + c] * invCount - mu * mu;
  float rs  = rsqrtf(var + EPSc);
  float sc  = g[c] * rs;
  ss[c]      = sc;
  ss[64 + c] = be[c] - mu * sc;
}

// ---------------------------------------------------------------------------
// GEMM1: [E,128](gathered, f16) @ [128,64] -> h1 (+bias), fused column stats.
// One wave -> T1 tiles of 16 edges x 64 cols; all 16 B tiles (128 VGPRs)
// resident in registers across the tile loop. Block = 8 waves = 512 edges.
// ---------------------------------------------------------------------------
__global__ __launch_bounds__(256) void edge_gemm1_kernel(
    const _Float16* __restrict__ xf16,
    const int* __restrict__ idx_i, const int* __restrict__ idx_j,
    const _Float16* __restrict__ Wp,   // packed [4nb][4kc][32lane][16]
    const float* __restrict__ bias,
    float* __restrict__ hout,
    float* __restrict__ stats)         // [64] sum, [64] sumsq
{
  __shared__ float lsum[64], lsq[64];
  int tid = threadIdx.x;
  if (tid < 64) { lsum[tid] = 0.0f; lsq[tid] = 0.0f; }
  __syncthreads();

  int wave = tid >> 5;
  int lane = tid & 31;
  int row  = lane & 15;
  int hi   = lane >> 4;
  int col0 = lane & 15;

  // Hoist all 16 weight tiles (nb*4+kc) into registers once per wave.
  v16h bmat[16];
  #pragma unroll
  for (int i = 0; i < 16; ++i)
    bmat[i] = *(const v16h*)(Wp + (((i << 5) + lane) << 4));

  float bcol[4];
  #pragma unroll
  for (int nb = 0; nb < 4; ++nb) bcol[nb] = bias[nb * 16 + col0];

  long wBase = (long)(blockIdx.x * 8 + wave) * (16 * T1);

  // Resolve all gather pointers up front and prefetch them so tiles 1..3
  // overlap their pointer-chase latency with tile 0's WMMA work.
  const _Float16* pis[T1];
  const _Float16* pjs[T1];
  #pragma unroll
  for (int t = 0; t < T1; ++t) {
    long e = wBase + t * 16 + row;
    pis[t] = xf16 + (long)idx_i[e] * 64;
    pjs[t] = xf16 + (long)idx_j[e] * 64;
  }
  #pragma unroll
  for (int t = 1; t < T1; ++t) {
    __builtin_prefetch(pis[t], 0, 0);   // global_prefetch_b8
    __builtin_prefetch(pjs[t], 0, 0);
  }

  v8f zero8 = {0.f,0.f,0.f,0.f,0.f,0.f,0.f,0.f};
  float s[4]  = {0.f,0.f,0.f,0.f};
  float sq[4] = {0.f,0.f,0.f,0.f};

  #pragma unroll
  for (int t = 0; t < T1; ++t) {
    long eBase = wBase + t * 16;
    const _Float16* pi = pis[t];
    const _Float16* pj = pjs[t];

    v8f acc[4] = {zero8, zero8, zero8, zero8};
    #pragma unroll
    for (int kc = 0; kc < 4; ++kc) {
      int k0 = (kc << 5) + (hi << 3);     // group0 K base (multiple of 8)
      int k1 = k0 + 16;                   // group1 K base
      const _Float16* s0 = (k0 < 64) ? (pi + k0) : (pj + (k0 - 64));
      const _Float16* s1 = (k1 < 64) ? (pi + k1) : (pj + (k1 - 64));
      v8h g0 = *(const v8h*)s0;           // two b128 gathers per lane
      v8h g1 = *(const v8h*)s1;
      v16h a;
      #pragma unroll
      for (int p = 0; p < 8; ++p) { a[p] = g0[p]; a[p + 8] = g1[p]; }
      #pragma unroll
      for (int nb = 0; nb < 4; ++nb) {
        acc[nb] = __builtin_amdgcn_wmma_f32_16x16x32_f16(
            false, a, false, bmat[nb * 4 + kc], (short)0, acc[nb], false, false);
      }
    }

    #pragma unroll
    for (int nb = 0; nb < 4; ++nb) {
      int col = nb * 16 + col0;
      #pragma unroll
      for (int r = 0; r < 8; ++r) {
        int m = r + 8 * hi;
        float v = acc[nb][r] + bcol[nb];
        hout[(eBase + m) * 64 + col] = v;
        s[nb] += v; sq[nb] += v * v;
      }
    }
  }

  #pragma unroll
  for (int nb = 0; nb < 4; ++nb) {
    int col = nb * 16 + col0;
    atomicAdd(&lsum[col], s[nb]);         // ds_add_f32
    atomicAdd(&lsq[col], sq[nb]);
  }
  __syncthreads();
  if (tid < 64) {
    atomicAdd(&stats[tid], lsum[tid]);
    atomicAdd(&stats[64 + tid], lsq[tid]);
  }
}

// ---------------------------------------------------------------------------
// GEMM over [R,64] fp32 rows @ [64,64]: optionally applies BN affine + ReLU
// to the input (fused into the A-load), adds bias, fused column stats.
// T1 tiles of 16 rows per wave; 8 B tiles resident in registers.
// ---------------------------------------------------------------------------
template <bool AFFINE>
__global__ __launch_bounds__(256) void rowmajor_gemm64_kernel(
    const float* __restrict__ in,      // [R,64]
    const float* __restrict__ ss,      // scale[64], shift[64]
    const _Float16* __restrict__ Wp,   // packed [4nb][2kc][32lane][16]
    const float* __restrict__ bias,
    float* __restrict__ out,
    float* __restrict__ stats, long R)
{
  __shared__ float lsum[64], lsq[64], lscale[64], lshift[64];
  int tid = threadIdx.x;
  if (tid < 64) {
    lsum[tid] = 0.0f; lsq[tid] = 0.0f;
    if (AFFINE) { lscale[tid] = ss[tid]; lshift[tid] = ss[64 + tid]; }
  }
  __syncthreads();

  int wave = tid >> 5;
  int lane = tid & 31;
  int row  = lane & 15;
  int hi   = lane >> 4;
  int col0 = lane & 15;

  v16h bmat[8];                           // 8 tiles (nb*2+kc), 64 VGPRs
  #pragma unroll
  for (int i = 0; i < 8; ++i)
    bmat[i] = *(const v16h*)(Wp + (((i << 5) + lane) << 4));

  float bcol[4];
  #pragma unroll
  for (int nb = 0; nb < 4; ++nb) bcol[nb] = bias[nb * 16 + col0];

  long wBase = (long)(blockIdx.x * 8 + wave) * (16 * T1);

  v8f zero8 = {0.f,0.f,0.f,0.f,0.f,0.f,0.f,0.f};
  float s[4]  = {0.f,0.f,0.f,0.f};
  float sq[4] = {0.f,0.f,0.f,0.f};

  #pragma unroll
  for (int t = 0; t < T1; ++t) {
    long rBase = wBase + t * 16;
    if (rBase >= R) break;                // R is a multiple of 16
    const float* prow = in + (rBase + row) * 64;

    v8f acc[4] = {zero8, zero8, zero8, zero8};
    #pragma unroll
    for (int kc = 0; kc < 2; ++kc) {
      v16h a;
      #pragma unroll
      for (int g = 0; g < 2; ++g) {
        int kb = (kc << 5) + (hi << 3) + g * 16;
        v4f f0 = *(const v4f*)(prow + kb);
        v4f f1 = *(const v4f*)(prow + kb + 4);
        #pragma unroll
        for (int p = 0; p < 4; ++p) {
          float va = f0[p], vb = f1[p];
          if (AFFINE) {
            va = fmaxf(va * lscale[kb + p]     + lshift[kb + p],     0.0f);
            vb = fmaxf(vb * lscale[kb + 4 + p] + lshift[kb + 4 + p], 0.0f);
          }
          a[g * 8 + p]     = (_Float16)va;
          a[g * 8 + 4 + p] = (_Float16)vb;
        }
      }
      #pragma unroll
      for (int nb = 0; nb < 4; ++nb) {
        acc[nb] = __builtin_amdgcn_wmma_f32_16x16x32_f16(
            false, a, false, bmat[nb * 2 + kc], (short)0, acc[nb], false, false);
      }
    }

    #pragma unroll
    for (int nb = 0; nb < 4; ++nb) {
      int col = nb * 16 + col0;
      #pragma unroll
      for (int r = 0; r < 8; ++r) {
        int m = r + 8 * hi;
        float v = acc[nb][r] + bcol[nb];
        out[(rBase + m) * 64 + col] = v;
        s[nb] += v; sq[nb] += v * v;
      }
    }
  }

  #pragma unroll
  for (int nb = 0; nb < 4; ++nb) {
    int col = nb * 16 + col0;
    atomicAdd(&lsum[col], s[nb]);
    atomicAdd(&lsq[col], sq[nb]);
  }
  __syncthreads();
  if (tid < 64) {
    atomicAdd(&stats[tid], lsum[tid]);
    atomicAdd(&stats[64 + tid], lsq[tid]);
  }
}

// m = relu(bn2(h2)); scatter-sum at idx_i
__global__ void scatter_kernel(const float* __restrict__ h2,
                               const float* __restrict__ ss,
                               const int* __restrict__ idx,
                               float* __restrict__ sout, long E) {
  long t = (long)blockIdx.x * blockDim.x + threadIdx.x;
  if (t >= E * 64) return;
  int c = (int)(t & 63);
  long e = t >> 6;
  float v = fmaxf(h2[t] * ss[c] + ss[64 + c], 0.0f);
  atomicAdd(&sout[(long)idx[e] * 64 + c], v);  // global_atomic_add_f32
}

// u = nodes + fi + fo (with degree normalization folded in)
__global__ void combine_kernel(const float* __restrict__ nodes,
                               const float* __restrict__ sfi,
                               const float* __restrict__ sfo,
                               const float* __restrict__ di,
                               const float* __restrict__ dj,
                               float* __restrict__ u, long n) {
  long t = (long)blockIdx.x * blockDim.x + threadIdx.x;
  if (t >= n) return;
  long nd = t >> 6;
  u[t] = nodes[t] + sfi[t] * di[nd] + sfo[t] * dj[nd];
}

// nodes += relu(bn(z)); refresh f16 mirror
__global__ void update_kernel(float* __restrict__ nodes,
                              _Float16* __restrict__ nodes16,
                              const float* __restrict__ z,
                              const float* __restrict__ ss, long n) {
  long t = (long)blockIdx.x * blockDim.x + threadIdx.x;
  if (t >= n) return;
  int c = (int)(t & 63);
  float v = fmaxf(z[t] * ss[c] + ss[64 + c], 0.0f);
  float nv = nodes[t] + v;
  nodes[t] = nv;
  nodes16[t] = (_Float16)nv;
}

__device__ inline void atomicMaxF(float* addr, float val) {
  unsigned int* ua = (unsigned int*)addr;
  unsigned int old = *ua;
  while (true) {
    float of = __uint_as_float(old);
    if (of >= val) break;
    unsigned int assumed = old;
    old = atomicCAS(ua, assumed, __float_as_uint(val));
    if (old == assumed) break;
  }
}

__global__ void segmax_kernel(const float* __restrict__ nodes,
                              const int* __restrict__ batch,
                              float* __restrict__ out, long n) {
  long t = (long)blockIdx.x * blockDim.x + threadIdx.x;
  if (t >= n) return;
  long nd = t >> 6;
  int c = (int)(t & 63);
  atomicMaxF(&out[(long)batch[nd] * 64 + c], nodes[t]);
}

// ---------------------------------------------------------------------------
// Host orchestration
// ---------------------------------------------------------------------------
extern "C" void kernel_launch(void* const* d_in, const int* in_sizes, int n_in,
                              void* d_out, int out_size, void* d_ws, size_t ws_size,
                              hipStream_t stream) {
  const float* x      = (const float*)d_in[0];
  const int*   ei     = (const int*)d_in[1];
  const int*   batch  = (const int*)d_in[2];
  const float* paW1 = (const float*)d_in[3],  *pa_b1 = (const float*)d_in[4];
  const float* pa_g1 = (const float*)d_in[5], *pa_be1 = (const float*)d_in[6];
  const float* paW2 = (const float*)d_in[7],  *pa_b2 = (const float*)d_in[8];
  const float* pa_g2 = (const float*)d_in[9], *pa_be2 = (const float*)d_in[10];
  const float* caW1 = (const float*)d_in[11], *ca_b1 = (const float*)d_in[12];
  const float* ca_g1 = (const float*)d_in[13], *ca_be1 = (const float*)d_in[14];
  const float* caW2 = (const float*)d_in[15], *ca_b2 = (const float*)d_in[16];
  const float* ca_g2 = (const float*)d_in[17], *ca_be2 = (const float*)d_in[18];
  const float* faW = (const float*)d_in[19],  *fa_b = (const float*)d_in[20];
  const float* fa_g = (const float*)d_in[21], *fa_be = (const float*)d_in[22];

  const int* src = ei;        // edge_index[0]
  const int* dst = ei + Ee;   // edge_index[1]

  // Workspace carve-up
  char* wsb = (char*)d_ws;
  size_t off = 0;
  auto take = [&](size_t bytes) -> char* {
    char* p = wsb + off;
    off += (bytes + 255) & ~(size_t)255;
    return p;
  };
  float*    h1       = (float*)take(Ee * 64 * 4);
  float*    h2       = (float*)take(Ee * 64 * 4);
  float*    nodes    = (float*)take(Nn * 64 * 4);
  _Float16* nodes16  = (_Float16*)take(Nn * 64 * 2);
  float*    sfi      = (float*)take(Nn * 64 * 4);
  float*    sfo      = (float*)take(Nn * 64 * 4);
  float*    u        = (float*)take(Nn * 64 * 4);
  float*    z        = (float*)take(Nn * 64 * 4);
  float*    dinv_dst = (float*)take(Nn * 4);
  float*    dinv_src = (float*)take(Nn * 4);
  float*    stats    = (float*)take(128 * 4);
  float*    ssbuf    = (float*)take(128 * 4);
  _Float16* paW1p    = (_Float16*)take(8192 * 2);
  _Float16* paW2p    = (_Float16*)take(4096 * 2);
  _Float16* caW1p    = (_Float16*)take(8192 * 2);
  _Float16* caW2p    = (_Float16*)take(4096 * 2);
  _Float16* faWp     = (_Float16*)take(4096 * 2);

  auto zero = [&](float* p, long n) {
    zero_f32_kernel<<<(unsigned)((n + 255) / 256), 256, 0, stream>>>(p, n);
  };

  // One-time per-launch prep
  pack_w_kernel<<<32, 256, 0, stream>>>(paW1, paW1p, 128);
  pack_w_kernel<<<16, 256, 0, stream>>>(paW2, paW2p, 64);
  pack_w_kernel<<<32, 256, 0, stream>>>(caW1, caW1p, 128);
  pack_w_kernel<<<16, 256, 0, stream>>>(caW2, caW2p, 64);
  pack_w_kernel<<<16, 256, 0, stream>>>(faW,  faWp,  64);
  nodes_init_kernel<<<(unsigned)((Nn * 64 + 255) / 256), 256, 0, stream>>>(
      x, nodes, nodes16, Nn * 64);

  zero(dinv_dst, Nn); zero(dinv_src, Nn);
  deg_count_kernel<<<(unsigned)((Ee + 255) / 256), 256, 0, stream>>>(dst, dinv_dst, Ee);
  deg_count_kernel<<<(unsigned)((Ee + 255) / 256), 256, 0, stream>>>(src, dinv_src, Ee);
  deg_inv_kernel<<<(unsigned)((Nn + 255) / 256), 256, 0, stream>>>(dinv_dst, Nn);
  deg_inv_kernel<<<(unsigned)((Nn + 255) / 256), 256, 0, stream>>>(dinv_src, Nn);

  const long rowsPerBlock = 8L * 16 * T1;                       // 512
  const unsigned edgeBlocks = (unsigned)(Ee / rowsPerBlock);    // 1000
  const unsigned nodeBlocks = (unsigned)((Nn + rowsPerBlock - 1) / rowsPerBlock); // 63
  const unsigned scatBlocks = (unsigned)((Ee * 64 + 255) / 256);

  auto run_edge_agg = [&](const int* ii, const int* ij,
                          const _Float16* W1p, const float* b1,
                          const float* g1, const float* be1,
                          const _Float16* W2p, const float* b2,
                          const float* g2, const float* be2, float* sout) {
    zero(stats, 128);
    edge_gemm1_kernel<<<edgeBlocks, 256, 0, stream>>>(
        nodes16, ii, ij, W1p, b1, h1, stats);
    bn_finalize_kernel<<<1, 64, 0, stream>>>(stats, g1, be1, 1.0f / (float)Ee, ssbuf);
    zero(stats, 128);
    rowmajor_gemm64_kernel<true><<<edgeBlocks, 256, 0, stream>>>(
        h1, ssbuf, W2p, b2, h2, stats, Ee);
    bn_finalize_kernel<<<1, 64, 0, stream>>>(stats, g2, be2, 1.0f / (float)Ee, ssbuf);
    zero(sout, Nn * 64);
    scatter_kernel<<<scatBlocks, 256, 0, stream>>>(h2, ssbuf, ii, sout, Ee);
  };

  for (int it = 0; it < ITERS; ++it) {
    // parent_agg: x_i = x[dst], x_j = x[src], scatter at dst
    run_edge_agg(dst, src, paW1p, pa_b1, pa_g1, pa_be1,
                 paW2p, pa_b2, pa_g2, pa_be2, sfi);
    // child_agg: x_i = x[src], x_j = x[dst], scatter at src
    run_edge_agg(src, dst, caW1p, ca_b1, ca_g1, ca_be1,
                 caW2p, ca_b2, ca_g2, ca_be2, sfo);
    combine_kernel<<<(unsigned)((Nn * 64 + 255) / 256), 256, 0, stream>>>(
        nodes, sfi, sfo, dinv_dst, dinv_src, u, Nn * 64);
    zero(stats, 128);
    rowmajor_gemm64_kernel<false><<<nodeBlocks, 256, 0, stream>>>(
        u, ssbuf, faWp, fa_b, z, stats, Nn);
    bn_finalize_kernel<<<1, 64, 0, stream>>>(stats, fa_g, fa_be, 1.0f / (float)Nn, ssbuf);
    update_kernel<<<(unsigned)((Nn * 64 + 255) / 256), 256, 0, stream>>>(
        nodes, nodes16, z, ssbuf, Nn * 64);
  }

  init_out_kernel<<<(unsigned)((out_size + 255) / 256), 256, 0, stream>>>(
      (float*)d_out, (long)out_size);
  segmax_kernel<<<(unsigned)((Nn * 64 + 255) / 256), 256, 0, stream>>>(
      nodes, batch, (float*)d_out, Nn * 64);
}
